// JointSinhLinearAttention_55052890800620
// MI455X (gfx1250) — matrix operand order repaired and checked by reading, not compiled, not run on hardware
//
#include <hip/hip_runtime.h>

typedef _Float16 v8h  __attribute__((ext_vector_type(8)));
typedef _Float16 v16h __attribute__((ext_vector_type(16)));
typedef float    v8f  __attribute__((ext_vector_type(8)));

#define NSEQ  2048
#define DH    64
#define BM    128      // Q rows per workgroup (8 waves x 16)
#define BN    32       // keys per iteration
#define WAVES 8
#define EPSV  1e-6f

static __device__ __forceinline__ v16h ld_frag(const _Float16* p0, const _Float16* p1) {
    union { v16h v; v8h h[2]; } u;
    u.h[0] = *(const v8h*)p0;   // ds_load_b128
    u.h[1] = *(const v8h*)p1;   // ds_load_b128
    return u.v;
}

// convert 8 consecutive f32 (two float4) to packed f16 with scale
static __device__ __forceinline__ v8h cvt8(float4 a, float4 b, float s) {
    v8h h;
    h[0] = (_Float16)(a.x * s); h[1] = (_Float16)(a.y * s);
    h[2] = (_Float16)(a.z * s); h[3] = (_Float16)(a.w * s);
    h[4] = (_Float16)(b.x * s); h[5] = (_Float16)(b.y * s);
    h[6] = (_Float16)(b.z * s); h[7] = (_Float16)(b.w * s);
    return h;
}

__global__ __launch_bounds__(256)
void sinh_attn_kernel(const float* __restrict__ Q,
                      const float* __restrict__ K,
                      const float* __restrict__ V,
                      const int*   __restrict__ mask,
                      float* __restrict__ out) {
    __shared__ __align__(16) _Float16 sQ[BM * DH];        // 16 KB, [row][d]
    __shared__ __align__(16) _Float16 sK[BN * DH];        //  4 KB, [key][d]
    __shared__ __align__(16) _Float16 sVT[DH * BN];       //  4 KB, [d][key]
    __shared__ __align__(16) _Float16 sP[WAVES * 16 * BN];//  8 KB, per-wave [m][key]

    const int tid  = threadIdx.x;
    const int wave = tid >> 5;
    const int lane = tid & 31;
    const int l15  = lane & 15;
    const int hi   = lane >> 4;          // 0: lanes 0-15, 1: lanes 16-31

    const int blk  = blockIdx.x;
    const int qblk = blk & ((NSEQ / BM) - 1);   // 16 q-blocks
    const int bh   = blk >> 4;                  // b*16 + h, 0..31
    const int b    = bh >> 4;                   // H = 16
    const long base = (long)bh * NSEQ * DH;
    const int q0   = qblk * BM;

    // ---- stage Q block: vectorized f32 -> f16, pre-scaled by D^-0.5 ----
    #pragma unroll
    for (int i = 0; i < (BM * DH) / (256 * 8); ++i) {     // 4 chunks/thread
        int c = tid + i * 256;
        int r = c >> 3, dc = (c & 7) * 8;
        const float* gq = &Q[base + (long)(q0 + r) * DH + dc];
        float4 a = *(const float4*)gq;
        float4 bq = *(const float4*)(gq + 4);
        *(v8h*)&sQ[r * DH + dc] = cvt8(a, bq, 0.125f);    // ds_store_b128
    }
    __syncthreads();

    // ---- load persistent Q A-fragments (16x32 f16, 2 d-tiles) ----
    const int qrow = wave * 16 + l15;
    v16h aq[2];
    #pragma unroll
    for (int dt = 0; dt < 2; ++dt) {
        const _Float16* p = &sQ[qrow * DH + dt * 32 + hi * 8];
        aq[dt] = ld_frag(p, p + 16);
    }

    const v8f vzero = {0.f, 0.f, 0.f, 0.f, 0.f, 0.f, 0.f, 0.f};
    v8f acc[4] = {vzero, vzero, vzero, vzero};   // O: 16 rows x 64 d (f32)
    v8f dsum   = vzero;                          // per-lane partial row sums

    _Float16* pwave = &sP[wave * (16 * BN)];

    // this thread's staging slot: one key x 8 consecutive d
    const int skey = tid >> 3;                   // 0..31
    const int sdc  = (tid & 7) * 8;              // 0..56

    // ---- register double-buffer: preload tile 0 ----
    float4 ka, kb2, va, vb2;
    float  mval;
    {
        long g = base + (long)skey * DH + sdc;
        mval = (mask[b * NSEQ + skey] != 0) ? 1.0f : 0.0f;
        ka  = *(const float4*)&K[g];
        kb2 = *(const float4*)&K[g + 4];
        va  = *(const float4*)&V[g];
        vb2 = *(const float4*)&V[g + 4];
    }

    const int ntiles = NSEQ / BN;                // 64
    for (int kt = 0; kt < ntiles; ++kt) {
        __syncthreads();                         // prior tile fully consumed

        // ---- commit staged registers into LDS (masked K, V^T) ----
        *(v8h*)&sK[skey * DH + sdc] = cvt8(ka, kb2, mval); // ds_store_b128
        {
            v8h vh = cvt8(va, vb2, 1.0f);
            #pragma unroll
            for (int j = 0; j < 8; ++j)                    // transpose into [d][key]
                sVT[(sdc + j) * BN + skey] = vh[j];
        }
        __syncthreads();

        // ---- issue global loads for tile kt+1 (overlap with compute) ----
        if (kt + 1 < ntiles) {
            long g = base + (long)((kt + 1) * BN + skey) * DH + sdc;
            mval = (mask[b * NSEQ + (kt + 1) * BN + skey] != 0) ? 1.0f : 0.0f;
            ka  = *(const float4*)&K[g];
            kb2 = *(const float4*)&K[g + 4];
            va  = *(const float4*)&V[g];
            vb2 = *(const float4*)&V[g + 4];
        }

        // ---- preload all K B-fragments (distinct regs -> pipelined DS) ----
        v16h bk[2][2];
        #pragma unroll
        for (int nt = 0; nt < 2; ++nt)
            #pragma unroll
            for (int dt = 0; dt < 2; ++dt) {
                const _Float16* kp = &sK[(nt * 16 + l15) * DH + dt * 32 + hi * 8];
                bk[nt][dt] = ld_frag(kp, kp + 16);
            }

        // ---- S = Q * K^T (2 key sub-tiles x K=64) ----
        v8f s[2];
        #pragma unroll
        for (int nt = 0; nt < 2; ++nt) {
            s[nt] = __builtin_amdgcn_wmma_f32_16x16x32_f16(
                        false, aq[0], false, bk[nt][0], (short)0, vzero, false, false);
            s[nt] = __builtin_amdgcn_wmma_f32_16x16x32_f16(
                        false, aq[1], false, bk[nt][1], (short)0, s[nt], false, false);
        }

        // ---- preload V B-fragments early (hide DS under the exp block) ----
        v16h bv[4];
        #pragma unroll
        for (int ot = 0; ot < 4; ++ot) {
            const _Float16* vp = &sVT[(ot * 16 + l15) * BN + hi * 8];
            bv[ot] = ld_frag(vp, vp + 16);
        }

        // ---- P = sinh(relu(S)); denom += rowsum(P); stage P for A-layout ----
        #pragma unroll
        for (int nt = 0; nt < 2; ++nt) {
            #pragma unroll
            for (int r = 0; r < 8; ++r) {
                float x = s[nt][r];
                float p = 0.0f;
                if (x > 0.0f) p = 0.5f * (__expf(x) - __expf(-x));  // sinh(relu(x))
                dsum[r] += p;
                int m = r + hi * 8;              // C layout -> row-major P in LDS
                pwave[m * BN + nt * 16 + l15] = (_Float16)p;
            }
        }

        // ---- O += P(16x32) * V(32x64): per-wave LDS round-trip for A-layout ----
        const _Float16* pp = &pwave[l15 * BN + hi * 8];  // DS in-order per wave
        v16h ap = ld_frag(pp, pp + 16);
        #pragma unroll
        for (int ot = 0; ot < 4; ++ot)
            acc[ot] = __builtin_amdgcn_wmma_f32_16x16x32_f16(
                          false, ap, false, bv[ot], (short)0, acc[ot], false, false);
    }

    // ---- reduce denominator across the 16 lanes of each half (wave32-safe) ----
    #pragma unroll
    for (int r = 0; r < 8; ++r) {
        float d = dsum[r];
        d += __shfl_xor(d, 1, 32);
        d += __shfl_xor(d, 2, 32);
        d += __shfl_xor(d, 4, 32);
        d += __shfl_xor(d, 8, 32);
        dsum[r] = 1.0f / fmaxf(d, EPSV);
    }

    // ---- normalize and store O ----
    #pragma unroll
    for (int ot = 0; ot < 4; ++ot) {
        #pragma unroll
        for (int r = 0; r < 8; ++r) {
            int m = r + hi * 8;
            out[base + (long)(q0 + wave * 16 + m) * DH + ot * 16 + l15] =
                acc[ot][r] * dsum[r];
        }
    }
}

extern "C" void kernel_launch(void* const* d_in, const int* in_sizes, int n_in,
                              void* d_out, int out_size, void* d_ws, size_t ws_size,
                              hipStream_t stream) {
    (void)in_sizes; (void)n_in; (void)out_size; (void)d_ws; (void)ws_size;
    const float* Q    = (const float*)d_in[0];
    const float* K    = (const float*)d_in[1];
    const float* V    = (const float*)d_in[2];
    const int*   mask = (const int*)d_in[3];
    float* out = (float*)d_out;

    const int B = 2, H = 16;
    dim3 grid(B * H * (NSEQ / BM));   // 512 workgroups
    dim3 block(256);                  // 8 wave32
    sinh_attn_kernel<<<grid, block, 0, stream>>>(Q, K, V, mask, out);
}
